// MultiHeadSelfAttention_60309930771175
// MI455X (gfx1250) — matrix operand order repaired and checked by reading, not compiled, not run on hardware
//
#include <hip/hip_runtime.h>

typedef __bf16 bf16;
typedef __attribute__((ext_vector_type(16))) __bf16 v16bf;
typedef __attribute__((ext_vector_type(8)))  __bf16 v8bf;
typedef __attribute__((ext_vector_type(8)))  float  v8f;
typedef unsigned int u32;
typedef __attribute__((ext_vector_type(4))) u32 u32x4;
typedef __attribute__((ext_vector_type(8))) u32 u32x8;

#define EMBED 512
#define SEQ   1024
#define NB    8
#define NH    8
#define HD    64

__device__ __forceinline__ v8f vzero8() {
    v8f z;
#pragma unroll
    for (int i = 0; i < 8; i++) z[i] = 0.0f;
    return z;
}

__device__ __forceinline__ v8f wmma_bf16(v16bf a, v16bf b, v8f c) {
    // D = A(16x32 bf16) * B(32x16 bf16) + C(16x16 f32)
    return __builtin_amdgcn_wmma_f32_16x16x32_bf16(
        false, a, false, b, (short)0, c, false, false);
}

// A-fragment (16x32, M = lane%16): contiguous k order (B uses the same), lane half hf:
//   elems 0..7  -> K = kb + hf*16 + e       (contiguous 8)
//   elems 8..15 -> K = kb + hf*16 + 8 + e   (contiguous 8)
template <typename P>
__device__ __forceinline__ v16bf load_a_frag(P rowp, int kb, int hf) {
    v8bf c0 = *(const v8bf*)(rowp + kb + hf * 16);
    v8bf c1 = *(const v8bf*)(rowp + kb + hf * 16 + 8);
    v16bf a;
#pragma unroll
    for (int i = 0; i < 8; i++) { a[i] = c0[i]; a[i + 8] = c1[i]; }
    return a;
}

// TDM: DMA a [rows x cols] bf16 tile (row stride `stride`, all in elements)
// from global to LDS. D# group0: count=1, lds_addr, 57b global_addr, type=2.
// D# group1: data_size=1 (2B), tensor dims == tile dims, dim0 stride.
__device__ __forceinline__ void tdm_load_2d(const bf16* gsrc, u32 ldsoff,
                                            u32 cols, u32 rows, u32 stride) {
    unsigned long long ga = (unsigned long long)(size_t)gsrc;
    u32x4 g0;
    g0[0] = 1u;                                              // count=1
    g0[1] = ldsoff;                                          // lds_addr (bytes)
    g0[2] = (u32)ga;                                         // global_addr[31:0]
    g0[3] = (u32)((ga >> 32) & 0x01FFFFFFu) | (2u << 30);    // addr[56:32] | type=2
    u32x8 g1;
    g1[0] = 0x00010000u;                                     // wg_mask=0, data_size=1 (2B)
    g1[1] = (cols & 0xFFFFu) << 16;                          // tensor_dim0 lo16
    g1[2] = ((cols >> 16) & 0xFFFFu) | ((rows & 0xFFFFu) << 16); // dim0 hi | dim1 lo
    g1[3] = ((rows >> 16) & 0xFFFFu) | (cols << 16);         // dim1 hi | tile_dim0
    g1[4] = rows & 0xFFFFu;                                  // tile_dim1 (tile_dim2=0)
    g1[5] = stride;                                          // tensor_dim0_stride lo32
    g1[6] = 0u;
    g1[7] = 0u;
    asm volatile("tensor_load_to_lds %0, %1" :: "s"(g0), "s"(g1) : "memory");
}

// ---------------------------------------------------------------- weights->bf16
__global__ __launch_bounds__(256) void wcvt_kernel(
    const float* __restrict__ wq, const float* __restrict__ wk,
    const float* __restrict__ wv, const float* __restrict__ wo,
    bf16* __restrict__ owq, bf16* __restrict__ owk,
    bf16* __restrict__ owv, bf16* __restrict__ owo) {
    int i = blockIdx.x * 256 + threadIdx.x;   // 0 .. 262143
    owq[i] = (bf16)wq[i];
    owk[i] = (bf16)wk[i];
    owv[i] = (bf16)wv[i];
    owo[i] = (bf16)wo[i];
}

// ---------------------------------------------------------------- layernorm + transpose to [B,S,C] bf16
__global__ __launch_bounds__(256) void ln_kernel(
    const float* __restrict__ x, const float* __restrict__ gamma,
    const float* __restrict__ beta, bf16* __restrict__ xn) {
    int lane = threadIdx.x & 31;
    int w = threadIdx.x >> 5;
    int g = blockIdx.x * 8 + w;          // (b,s) index
    int b = g >> 10;
    int s = g & 1023;
    const float* xb = x + (size_t)b * EMBED * SEQ;
    float vals[16];
    float sum = 0.0f, sq = 0.0f;
#pragma unroll
    for (int i = 0; i < 16; i++) {
        int c = i * 32 + lane;
        float v = xb[(size_t)c * SEQ + s];
        vals[i] = v; sum += v; sq += v * v;
    }
#pragma unroll
    for (int off = 16; off >= 1; off >>= 1) {
        sum += __shfl_xor(sum, off, 32);
        sq  += __shfl_xor(sq,  off, 32);
    }
    float mu  = sum * (1.0f / EMBED);
    float var = sq * (1.0f / EMBED) - mu * mu;
    float rs  = rsqrtf(var + 1e-5f);
    bf16* dst = xn + ((size_t)(b * SEQ + s)) * EMBED;
#pragma unroll
    for (int i = 0; i < 16; i++) {
        int c = i * 32 + lane;
        dst[c] = (bf16)((vals[i] - mu) * rs * gamma[c] + beta[c]);
    }
}

// ---------------------------------------------------------------- QKV projection
// Block: 8 waves sharing one TDM-staged 16-row A-tile; wave w owns N-strip w*64.
// Even/odd two-stage register pipeline: no buffer-rotation copies.
__global__ __launch_bounds__(256) void qkv_kernel(
    const bf16* __restrict__ xn,
    const bf16* __restrict__ wq, const bf16* __restrict__ wk, const bf16* __restrict__ wv,
    const float* __restrict__ bq, const float* __restrict__ bk, const float* __restrict__ bv,
    bf16* __restrict__ qo, bf16* __restrict__ ko, bf16* __restrict__ vt) {
    __shared__ __align__(128) bf16 atile[16 * EMBED];   // 16KB
    int lane = threadIdx.x & 31;
    int w = threadIdx.x >> 5;
    int lane16 = lane & 15;
    int hf = lane >> 4;
    int blk = blockIdx.x;                    // 0..1535
    int m0   = (blk & 63) * 16;
    int b    = (blk >> 6) & 7;
    int wsel = blk >> 9;                     // 0=Q 1=K 2=V
    int n0   = w * 64;
    const bf16* W    = (wsel == 0) ? wq : ((wsel == 1) ? wk : wv);
    const float* bia = (wsel == 0) ? bq : ((wsel == 1) ? bk : bv);

    if (w == 0) {
        tdm_load_2d(xn + (size_t)(b * SEQ + m0) * EMBED, (u32)(size_t)atile,
                    EMBED, 16, EMBED);
        __builtin_amdgcn_s_wait_tensorcnt(0);
    }
    __syncthreads();

    const bf16* arow = atile + lane16 * EMBED;   // LDS row for this lane's M
    const bf16* wrow[4];
#pragma unroll
    for (int ns = 0; ns < 4; ns++)
        wrow[ns] = W + (size_t)(n0 + ns * 16 + lane16) * EMBED + hf * 16;

    v8f acc[4];
#pragma unroll
    for (int ns = 0; ns < 4; ns++) acc[ns] = vzero8();

    v16bf aA = load_a_frag(arow, 0, hf);
    v16bf bA[4];
#pragma unroll
    for (int ns = 0; ns < 4; ns++) bA[ns] = *(const v16bf*)(wrow[ns]);

    for (int kb = 0; kb < EMBED; kb += 64) {
        v16bf aB = load_a_frag(arow, kb + 32, hf);
        v16bf bB[4];
#pragma unroll
        for (int ns = 0; ns < 4; ns++) bB[ns] = *(const v16bf*)(wrow[ns] + kb + 32);
#pragma unroll
        for (int ns = 0; ns < 4; ns++) acc[ns] = wmma_bf16(aA, bA[ns], acc[ns]);
        int kn = (kb + 64) & (EMBED - 1);    // wrapped prefetch (last one unused)
        aA = load_a_frag(arow, kn, hf);
#pragma unroll
        for (int ns = 0; ns < 4; ns++) bA[ns] = *(const v16bf*)(wrow[ns] + kn);
#pragma unroll
        for (int ns = 0; ns < 4; ns++) acc[ns] = wmma_bf16(aB, bB[ns], acc[ns]);
    }

    float scale = (wsel == 0) ? 0.125f : 1.0f;   // fold 1/sqrt(64) into Q
#pragma unroll
    for (int ns = 0; ns < 4; ns++) {
        int N = n0 + ns * 16 + lane16;
        float bb = bia[N];
        if (wsel < 2) {
            bf16* dst = (wsel == 0) ? qo : ko;
#pragma unroll
            for (int r = 0; r < 8; r++) {
                int s = m0 + r + 8 * hf;          // D-layout: M = r + 8*half
                dst[((size_t)(b * SEQ + s)) * EMBED + N] = (bf16)((acc[ns][r] + bb) * scale);
            }
        } else {
            // V stored transposed: vt[b][h][d][s], r-consecutive == s-consecutive
            v8bf pv;
#pragma unroll
            for (int r = 0; r < 8; r++) pv[r] = (bf16)(acc[ns][r] + bb);
            int hh = N >> 6, dd = N & 63;
            *(v8bf*)(vt + ((size_t)((b * NH + hh) * HD + dd)) * SEQ + m0 + 8 * hf) = pv;
        }
    }
}

// ---------------------------------------------------------------- flash attention
// Workgroup = (b, h, 128 q-rows). Entire per-head K (1024x64) and V^T (64x1024)
// are TDM-DMA'd into LDS once (256KB of the 320KB WGP LDS); the whole k-loop
// then runs from LDS with zero global traffic.
#define ATTN_KLDS   (SEQ * HD * 2)                    // 131072 B
#define ATTN_VLDS   (HD * SEQ * 2)                    // 131072 B
#define ATTN_SCOFF  (ATTN_KLDS + ATTN_VLDS)           // 262144
#define ATTN_PBOFF  (ATTN_SCOFF + 8 * 16 * 33 * 4)    // 279040 (64B aligned)
#define ATTN_CROFF  (ATTN_PBOFF + 8 * 16 * 32 * 2)    // 287232
#define ATTN_LROFF  (ATTN_CROFF + 8 * 16 * 4)         // 287744
#define ATTN_SMEM   (ATTN_LROFF + 8 * 16 * 4)         // 288256 B (< 320KB)

__global__ __launch_bounds__(256) void attn_kernel(
    const bf16* __restrict__ q, const bf16* __restrict__ k,
    const bf16* __restrict__ vt, bf16* __restrict__ ao) {
    extern __shared__ __align__(128) char smem[];
    bf16*  ktile = (bf16*)smem;                       // [1024][64]
    bf16*  vtile = (bf16*)(smem + ATTN_KLDS);         // [64][1024]
    float* scb   = (float*)(smem + ATTN_SCOFF);       // [8][16][33]
    bf16*  pbb   = (bf16*)(smem + ATTN_PBOFF);        // [8][16][32]
    float* corrb = (float*)(smem + ATTN_CROFF);       // [8][16]
    float* lrowb = (float*)(smem + ATTN_LROFF);       // [8][16]

    int lane = threadIdx.x & 31;
    int w = threadIdx.x >> 5;
    int lane16 = lane & 15;
    int hf = lane >> 4;
    int blk = blockIdx.x;                 // 0..511
    int qt = blk & 7;
    int h  = (blk >> 3) & 7;
    int b  = blk >> 6;
    int q0 = qt * 128 + w * 16;

    // stage K and V^T for this (b,h) into LDS via TDM (one wave issues)
    if (w == 0) {
        tdm_load_2d(k + (size_t)(b * SEQ) * EMBED + h * HD, (u32)(size_t)ktile,
                    HD, SEQ, EMBED);                  // 1024 rows x 64 cols, stride 512
        tdm_load_2d(vt + (size_t)((b * NH + h) * HD) * SEQ, (u32)(size_t)vtile,
                    SEQ, HD, SEQ);                    // 64 rows x 1024 cols, stride 1024
        __builtin_amdgcn_s_wait_tensorcnt(0);
    }

    const bf16* qrow = q + ((size_t)(b * SEQ + q0 + lane16)) * EMBED + h * HD;
    v16bf qa0 = load_a_frag(qrow, 0, hf);
    v16bf qa1 = load_a_frag(qrow, 32, hf);

    __syncthreads();

    v8f acc[4];
#pragma unroll
    for (int dc = 0; dc < 4; dc++) acc[dc] = vzero8();
    float m = -1e30f, l = 0.0f;
    float* scw = scb + (size_t)w * 16 * 33;
    bf16*  pbw = pbb + (size_t)w * 16 * 32;

    for (int kb = 0; kb < SEQ; kb += 32) {
        // K/V fragments from LDS
        const bf16* kr0 = ktile + (size_t)(kb + lane16) * HD;
        const bf16* kr1 = kr0 + (size_t)16 * HD;
        v16bf kb00 = *(const v16bf*)(kr0 + hf * 16);
        v16bf kb01 = *(const v16bf*)(kr0 + 32 + hf * 16);
        v16bf kb10 = *(const v16bf*)(kr1 + hf * 16);
        v16bf kb11 = *(const v16bf*)(kr1 + 32 + hf * 16);
        v16bf vb[4];
#pragma unroll
        for (int dc = 0; dc < 4; dc++)
            vb[dc] = *(const v16bf*)(vtile + (size_t)(dc * 16 + lane16) * SEQ + kb + hf * 16);

        v8f s0 = vzero8(), s1 = vzero8();
        s0 = wmma_bf16(qa0, kb00, s0);
        s0 = wmma_bf16(qa1, kb01, s0);
        s1 = wmma_bf16(qa0, kb10, s1);
        s1 = wmma_bf16(qa1, kb11, s1);

        // D-layout -> LDS (per-wave private region, same-wave ordering only)
#pragma unroll
        for (int r = 0; r < 8; r++) {
            scw[(r + 8 * hf) * 33 + lane16]      = s0[r];
            scw[(r + 8 * hf) * 33 + 16 + lane16] = s1[r];
        }
        asm volatile("s_wait_dscnt 0" ::: "memory");
        // online softmax: lane owns (row = lane%16, 16-column half = lane/16)
        {
            int row = lane16;
            int co = hf * 16;
            float mx = -1e30f;
#pragma unroll
            for (int j = 0; j < 16; j++) mx = fmaxf(mx, scw[row * 33 + co + j]);
            mx = fmaxf(mx, __shfl_xor(mx, 16, 32));
            float mnew = fmaxf(m, mx);
            float cf = __expf(m - mnew);
            float sum = 0.0f;
#pragma unroll
            for (int j = 0; j < 16; j++) {
                float p = __expf(scw[row * 33 + co + j] - mnew);
                sum += p;
                pbw[row * 32 + co + j] = (bf16)p;
            }
            sum += __shfl_xor(sum, 16, 32);
            l = l * cf + sum;
            m = mnew;
            if (lane < 16) corrb[w * 16 + row] = cf;
        }
        asm volatile("s_wait_dscnt 0" ::: "memory");
        // rescale running output accumulators (per-row factor)
        float cr[8];
#pragma unroll
        for (int r = 0; r < 8; r++) cr[r] = corrb[w * 16 + r + 8 * hf];
#pragma unroll
        for (int dc = 0; dc < 4; dc++)
#pragma unroll
            for (int r = 0; r < 8; r++) acc[dc][r] *= cr[r];
        // P as A-fragment from LDS
        v16bf pa;
        {
            v8bf p0 = *(const v8bf*)&pbw[lane16 * 32 + hf * 16];
            v8bf p1 = *(const v8bf*)&pbw[lane16 * 32 + hf * 16 + 8];
#pragma unroll
            for (int i = 0; i < 8; i++) { pa[i] = p0[i]; pa[i + 8] = p1[i]; }
        }
        // out += P(16x32) . V(32x64)
#pragma unroll
        for (int dc = 0; dc < 4; dc++)
            acc[dc] = wmma_bf16(pa, vb[dc], acc[dc]);
    }

    if (lane < 16) lrowb[w * 16 + lane16] = l;
    asm volatile("s_wait_dscnt 0" ::: "memory");
    float li[8];
#pragma unroll
    for (int r = 0; r < 8; r++) li[r] = 1.0f / lrowb[w * 16 + r + 8 * hf];
#pragma unroll
    for (int dc = 0; dc < 4; dc++)
#pragma unroll
        for (int r = 0; r < 8; r++) {
            int s = q0 + r + 8 * hf;
            ao[((size_t)(b * SEQ + s)) * EMBED + h * HD + dc * 16 + lane16] =
                (bf16)(acc[dc][r] * li[r]);
        }
}

// ---------------------------------------------------------------- out-proj + bias + residual + transpose
__global__ __launch_bounds__(256) void proj_kernel(
    const bf16* __restrict__ ao, const bf16* __restrict__ wo,
    const float* __restrict__ bo, const float* __restrict__ x,
    float* __restrict__ out) {
    __shared__ __align__(128) bf16 atile[16 * EMBED];   // 16KB
    int lane = threadIdx.x & 31;
    int w = threadIdx.x >> 5;
    int lane16 = lane & 15;
    int hf = lane >> 4;
    int blk = blockIdx.x;                // 0..511
    int m0 = (blk & 63) * 16;
    int b  = blk >> 6;
    int n0 = w * 64;

    if (w == 0) {
        tdm_load_2d(ao + (size_t)(b * SEQ + m0) * EMBED, (u32)(size_t)atile,
                    EMBED, 16, EMBED);
        __builtin_amdgcn_s_wait_tensorcnt(0);
    }
    __syncthreads();

    const bf16* arow = atile + lane16 * EMBED;
    const bf16* wrow[4];
#pragma unroll
    for (int ns = 0; ns < 4; ns++)
        wrow[ns] = wo + (size_t)(n0 + ns * 16 + lane16) * EMBED + hf * 16;

    v8f acc[4];
#pragma unroll
    for (int ns = 0; ns < 4; ns++) acc[ns] = vzero8();

    v16bf aA = load_a_frag(arow, 0, hf);
    v16bf bA[4];
#pragma unroll
    for (int ns = 0; ns < 4; ns++) bA[ns] = *(const v16bf*)(wrow[ns]);

    for (int kb = 0; kb < EMBED; kb += 64) {
        v16bf aB = load_a_frag(arow, kb + 32, hf);
        v16bf bB[4];
#pragma unroll
        for (int ns = 0; ns < 4; ns++) bB[ns] = *(const v16bf*)(wrow[ns] + kb + 32);
#pragma unroll
        for (int ns = 0; ns < 4; ns++) acc[ns] = wmma_bf16(aA, bA[ns], acc[ns]);
        int kn = (kb + 64) & (EMBED - 1);
        aA = load_a_frag(arow, kn, hf);
#pragma unroll
        for (int ns = 0; ns < 4; ns++) bA[ns] = *(const v16bf*)(wrow[ns] + kn);
#pragma unroll
        for (int ns = 0; ns < 4; ns++) acc[ns] = wmma_bf16(aB, bB[ns], acc[ns]);
    }
#pragma unroll
    for (int ns = 0; ns < 4; ns++) {
        int N = n0 + ns * 16 + lane16;
        float bb = bo[N];
        size_t base = ((size_t)(b * EMBED + N)) * SEQ + m0 + 8 * hf; // r-consecutive == s-consecutive
        v8f xv = *(const v8f*)(x + base);
        v8f y;
#pragma unroll
        for (int r = 0; r < 8; r++) y[r] = acc[ns][r] + bb + xv[r];
        *(v8f*)(out + base) = y;
    }
}

extern "C" void kernel_launch(void* const* d_in, const int* in_sizes, int n_in,
                              void* d_out, int out_size, void* d_ws, size_t ws_size,
                              hipStream_t stream) {
    const float* x     = (const float*)d_in[0];
    const float* Wq    = (const float*)d_in[1];
    const float* bq    = (const float*)d_in[2];
    const float* Wk    = (const float*)d_in[3];
    const float* bk    = (const float*)d_in[4];
    const float* Wv    = (const float*)d_in[5];
    const float* bv    = (const float*)d_in[6];
    const float* Wo    = (const float*)d_in[7];
    const float* bo    = (const float*)d_in[8];
    const float* gamma = (const float*)d_in[9];
    const float* beta  = (const float*)d_in[10];
    float* out = (float*)d_out;

    const size_t ACT = (size_t)NB * SEQ * EMBED;   // 4 Mi elems
    const size_t WSZ = (size_t)EMBED * EMBED;      // 256 Ki elems
    bf16* xn  = (bf16*)d_ws;
    bf16* q   = xn + ACT;
    bf16* k   = q + ACT;
    bf16* vt  = k + ACT;          // [B][H][64][S]
    bf16* ao  = vt + ACT;
    bf16* bwq = ao + ACT;
    bf16* bwk = bwq + WSZ;
    bf16* bwv = bwk + WSZ;
    bf16* bwo = bwv + WSZ;

    wcvt_kernel<<<1024, 256, 0, stream>>>(Wq, Wk, Wv, Wo, bwq, bwk, bwv, bwo);
    ln_kernel<<<1024, 256, 0, stream>>>(x, gamma, beta, xn);
    qkv_kernel<<<1536, 256, 0, stream>>>(xn, bwq, bwk, bwv, bq, bk, bv, q, k, vt);
    attn_kernel<<<512, 256, ATTN_SMEM, stream>>>(q, k, vt, ao);
    proj_kernel<<<512, 256, 0, stream>>>(ao, bwo, bo, x, out);
}
